// Self_Attention_55525337203375
// MI455X (gfx1250) — matrix operand order repaired and checked by reading, not compiled, not run on hardware
//
#include <hip/hip_runtime.h>

// Self-attention, b=32, seq=2048, head_dim=32, fp32 I/O.
// Kernel 1: QKV projection via v_wmma_f32_16x16x32_f16 ->
//   Q row-major f16 (B-frag source), K/V PRE-SWIZZLED so kernel 2's
//   A-fragments are contiguous 32B/lane loads.
// Kernel 2: flash attention on transposed tiles (S^T = K*Q^T, O^T = V^T*P^T):
//   key-axis softmax is in-lane; 64-key blocks; all exponentials are raw
//   v_exp_f32 (log2e folded into the Q scale). No LDS in the hot loop.
// ws usage: 12 MB (L2-resident at 192 MB).

typedef __attribute__((ext_vector_type(16))) _Float16 v16h;
typedef __attribute__((ext_vector_type(8)))  float    v8f;

#define SEQ 2048
#define HD  32
// (1/sqrt(32)) * log2(e): scores produced directly in log2 domain
#define QK_SCALE_LOG2E 0.25504167f

#if __has_builtin(__builtin_amdgcn_exp2f)
#define EXP2(x) __builtin_amdgcn_exp2f(x)
#else
#define EXP2(x) exp2f(x)
#endif

__device__ __forceinline__ v8f wmma16x16x32(v16h a, v16h b, v8f c) {
  return __builtin_amdgcn_wmma_f32_16x16x32_f16(false, a, false, b,
                                                (short)0, c, false, false);
}

// swizzle for A-fragment-friendly storage of a 32-wide axis:
// memory order [0..31] (contiguous)
__device__ __forceinline__ int swz32(int c) {
  return c;
}

// A-fragment K index pattern (contiguous: k = half*16 + 2p)
__device__ __forceinline__ int a_frag_k(int p, int half) {
  return 2 * p + half * 16;
}

__device__ __forceinline__ unsigned pkh(float a, float b) {
  union { unsigned u; _Float16 h[2]; } r;
  r.h[0] = (_Float16)a; r.h[1] = (_Float16)b;
  return r.u;
}

// ---------------- Kernel 1: QKV projection -----------------------------
__global__ void __launch_bounds__(256)
qkv_project(const float* __restrict__ x, const float* __restrict__ w,
            _Float16* __restrict__ Q, _Float16* __restrict__ Ks,
            _Float16* __restrict__ Vs) {
  const int lane = threadIdx.x & 31;
  const int wave = threadIdx.x >> 5;
  const int tile = blockIdx.x * 8 + wave;   // 32 batches * 128 tiles
  const int b    = tile >> 7;
  const int row0 = (tile & 127) * 16;
  const int half = lane >> 4;
  const int n    = lane & 15;

  // Weight B-fragments: lane = output col, K = half*16..+15 contiguous
  v16h bw[6];
#pragma unroll
  for (int m = 0; m < 3; ++m) {
    const float* W = w + m * HD * HD;
#pragma unroll
    for (int h = 0; h < 2; ++h) {
      v16h f;
#pragma unroll
      for (int kk = 0; kk < 16; ++kk)
        f[kk] = (_Float16)W[(half * 16 + kk) * HD + (h * 16 + n)];
      bw[m * 2 + h] = f;
    }
  }

  // X A-fragment
  const float* xb = x + (size_t)b * (SEQ * HD);
  v16h a;
#pragma unroll
  for (int p = 0; p < 8; ++p) {
    const int k = a_frag_k(p, half);
    const float2 xv = *(const float2*)(xb + (row0 + n) * HD + k);
    a[2 * p]     = (_Float16)xv.x;
    a[2 * p + 1] = (_Float16)xv.y;
  }

  const v8f z = {};
  v8f q0 = wmma16x16x32(a, bw[0], z);
  v8f q1 = wmma16x16x32(a, bw[1], z);
  v8f k0 = wmma16x16x32(a, bw[2], z);
  v8f k1 = wmma16x16x32(a, bw[3], z);
  v8f v0 = wmma16x16x32(a, bw[4], z);
  v8f v1 = wmma16x16x32(a, bw[5], z);

  _Float16* Qb = Q  + (size_t)b * (SEQ * HD);
  _Float16* Kb = Ks + (size_t)b * (SEQ * HD);
  _Float16* Vb = Vs + (size_t)b * (HD * SEQ);
  const int kc0 = swz32(n);
  const int kc1 = swz32(n + 16);
#pragma unroll
  for (int v = 0; v < 8; ++v) {
    const int row = row0 + v + half * 8;
    Qb[row * HD + n]       = (_Float16)q0[v];
    Qb[row * HD + n + 16]  = (_Float16)q1[v];
    Kb[row * HD + kc0]     = (_Float16)k0[v];
    Kb[row * HD + kc1]     = (_Float16)k1[v];
    const int ks = swz32(row & 31);
    const int kb = row >> 5;
    Vb[n * SEQ + kb * 32 + ks]        = (_Float16)v0[v];
    Vb[(n + 16) * SEQ + kb * 32 + ks] = (_Float16)v1[v];
  }
}

// ---------------- Kernel 2: flash attention (transposed S) -------------
__global__ void __launch_bounds__(256)
flash_attn(const _Float16* __restrict__ Q, const _Float16* __restrict__ Ks,
           const _Float16* __restrict__ Vs, float* __restrict__ out) {
  const int lane = threadIdx.x & 31;
  const int wave = threadIdx.x >> 5;
  const int tile = blockIdx.x * 8 + wave;
  const int b    = tile >> 7;
  const int row0 = (tile & 127) * 16;
  const int half = lane >> 4;
  const int n    = lane & 15;

  const _Float16* Qb = Q  + (size_t)b * (SEQ * HD);
  const _Float16* Kb = Ks + (size_t)b * (SEQ * HD);
  const _Float16* Vb = Vs + (size_t)b * (HD * SEQ);

  // Q^T B-fragment; fold 1/sqrt(HD) * log2(e) once
  v16h qb;
  {
    const v16h qraw = *(const v16h*)(Qb + (row0 + n) * HD + half * 16);
#pragma unroll
    for (int i = 0; i < 16; ++i)
      qb[i] = (_Float16)((float)qraw[i] * QK_SCALE_LOG2E);
  }

  v8f o0 = {}, o1 = {};                 // O^T tiles: comps 0..15 / 16..31
  float mrun = -3.0e38f, lrun = 0.0f;   // per-query, log2 domain
  const v8f z = {};

  for (int j = 0; j < SEQ; j += 64) {
    // K A-fragments for 4 key tiles (pre-swizzled -> contiguous)
    v16h ka[4];
#pragma unroll
    for (int t = 0; t < 4; ++t)
      ka[t] = *(const v16h*)(Kb + (j + t * 16 + n) * HD + half * 16);

    v8f s[4];
#pragma unroll
    for (int t = 0; t < 4; ++t)
      s[t] = wmma16x16x32(ka[t], qb, z);   // S^T: rows=keys, cols=queries

    // in-lane max over the 32 key-scores this lane holds
    float mx = s[0][0];
#pragma unroll
    for (int t = 0; t < 4; ++t)
#pragma unroll
      for (int v = 0; v < 8; ++v) mx = fmaxf(mx, s[t][v]);
    mx = fmaxf(mx, __shfl_xor(mx, 16));       // combine lane pair
    const float mnew  = fmaxf(mrun, mx);
    const float alpha = EXP2(mrun - mnew);

    float p[4][8], rs = 0.0f;
#pragma unroll
    for (int t = 0; t < 4; ++t)
#pragma unroll
      for (int v = 0; v < 8; ++v) {
        p[t][v] = EXP2(s[t][v] - mnew);       // raw v_exp_f32
        rs += p[t][v];
      }
    rs += __shfl_xor(rs, 16);
    lrun = lrun * alpha + rs;
    mrun = mnew;
#pragma unroll
    for (int v = 0; v < 8; ++v) { o0[v] *= alpha; o1[v] *= alpha; }

    // P^T B-fragments: own half supplies 8 keys, partner lane (xor 16) the
    // other 8; exchanged as packed-f16 dwords.
    unsigned pt[4][4], ot[4][4];
#pragma unroll
    for (int t = 0; t < 4; ++t)
#pragma unroll
      for (int i = 0; i < 4; ++i)
        pt[t][i] = pkh(p[t][2 * i], p[t][2 * i + 1]);
#pragma unroll
    for (int t = 0; t < 4; ++t)
#pragma unroll
      for (int i = 0; i < 4; ++i)
        ot[t][i] = (unsigned)__shfl_xor((int)pt[t][i], 16);

    union { v16h h; unsigned u[8]; } pbA, pbB;
#pragma unroll
    for (int i = 0; i < 4; ++i) {
      pbA.u[i]     = half ? ot[1][i] : pt[0][i];  // keys j+0..7  | j+16..23
      pbA.u[4 + i] = half ? pt[1][i] : ot[0][i];  // keys j+8..15 | j+24..31
      pbB.u[i]     = half ? ot[3][i] : pt[2][i];  // keys j+32..  | j+48..
      pbB.u[4 + i] = half ? pt[3][i] : ot[2][i];
    }

    // V^T A-fragments (pre-swizzled key axis -> contiguous)
    const v16h va0 = *(const v16h*)(Vb + n * SEQ + j + half * 16);
    const v16h va1 = *(const v16h*)(Vb + (n + 16) * SEQ + j + half * 16);
    const v16h va2 = *(const v16h*)(Vb + n * SEQ + j + 32 + half * 16);
    const v16h va3 = *(const v16h*)(Vb + (n + 16) * SEQ + j + 32 + half * 16);
    o0 = wmma16x16x32(va0, pbA.h, o0);
    o1 = wmma16x16x32(va1, pbA.h, o1);
    o0 = wmma16x16x32(va2, pbB.h, o0);
    o1 = wmma16x16x32(va3, pbB.h, o1);
  }

  // O^T: lane holds O[comp][query]; query = row0+n, comp = v+8*half (+16)
  float* ob = out + (size_t)b * (SEQ * HD);
  const float inv = 1.0f / lrun;
  const int qrow = row0 + n;
#pragma unroll
  for (int v = 0; v < 8; ++v) {
    ob[qrow * HD + v + 8 * half]      = o0[v] * inv;
    ob[qrow * HD + 16 + v + 8 * half] = o1[v] * inv;
  }
}

// ---------------- launch ----------------------------------------------
extern "C" void kernel_launch(void* const* d_in, const int* in_sizes, int n_in,
                              void* d_out, int out_size, void* d_ws, size_t ws_size,
                              hipStream_t stream) {
  (void)in_sizes; (void)n_in; (void)out_size; (void)ws_size;
  const float* x = (const float*)d_in[0];   // [32, 65536] f32
  const float* w = (const float*)d_in[1];   // [3, 32, 32] f32
  float* out = (float*)d_out;               // [32, 65536] f32

  const size_t per = (size_t)32 * SEQ * HD; // 2,097,152 f16 elems each
  _Float16* Q  = (_Float16*)d_ws;
  _Float16* Ks = Q + per;
  _Float16* Vs = Ks + per;                  // 12 MB of ws

  qkv_project<<<512, 256, 0, stream>>>(x, w, Q, Ks, Vs);
  flash_attn<<<512, 256, 0, stream>>>(Q, Ks, Vs, out);
}
